// SRU_Formula_Cell_64845416235481
// MI455X (gfx1250) — compile-verified, matrix operand order checked by reading, not run on hardware
//
#include <hip/hip_runtime.h>
#include <hip/hip_bf16.h>
#include <stdint.h>
#include <stddef.h>

// ---- problem constants (match reference) ----
#define N_IN_C   1024
#define N_OUT_C  512
#define T_STEPS  512
#define BATCH    32
#define M_TOT    (T_STEPS * BATCH)   // 16384
#define N_PROJ   (4 * N_OUT_C)       // 2048 (x | f | r | cx concatenated)

typedef __attribute__((ext_vector_type(16))) __bf16 v16bf;
typedef __attribute__((ext_vector_type(8)))  __bf16 v8bf;
typedef __attribute__((ext_vector_type(8)))  float  v8f;
typedef __attribute__((ext_vector_type(4)))  int    i4vec;

typedef __attribute__((address_space(1))) i4vec* gptr128;   // global int4*
typedef __attribute__((address_space(3))) i4vec* lptr128;   // LDS int4*

// ---- GEMM tiling ----
#define BM 128
#define BN 128
#define BK 32
#define LDS_K 40            // 32 elems + 8 elems (16B) pad -> 80B row stride, 16B aligned

// ---- CDNA5 async global->LDS path (guarded; falls back to load+ds_store) ----
#if defined(__has_builtin)
#if __has_builtin(__builtin_amdgcn_global_load_async_to_lds_b128) && \
    __has_builtin(__builtin_amdgcn_s_wait_asynccnt)
#define HAVE_ASYNC_LDS 1
#endif
#endif
#ifndef HAVE_ASYNC_LDS
#define HAVE_ASYNC_LDS 0
#endif

__device__ __forceinline__ void copy16_to_lds(const __bf16* __restrict__ g,
                                              __bf16* __restrict__ l) {
#if HAVE_ASYNC_LDS
    __builtin_amdgcn_global_load_async_to_lds_b128(
        (gptr128)g, (lptr128)l, 0 /*imm offset*/, 0 /*cpol*/);
#else
    __builtin_prefetch((const void*)(g + 64), 0, 3);
    *(uint4*)l = *(const uint4*)g;
#endif
}

__device__ __forceinline__ void wait_tile_copies() {
#if HAVE_ASYNC_LDS
    __builtin_amdgcn_s_wait_asynccnt(0);   // async writes to LDS landed
#endif
    // (sync fallback: __syncthreads() below inserts s_wait_dscnt)
}

// ============================================================
// elementwise fp32 -> bf16 conversion
// ============================================================
__global__ void cvt_f32_bf16(const float* __restrict__ src, __bf16* __restrict__ dst, int n) {
    int i = blockIdx.x * blockDim.x + threadIdx.x;
    if (i < n) dst[i] = (__bf16)src[i];
}

// build concatenated bias [2048]: [0 | b_f | b_r | b_cx]
__global__ void build_bias(const float* __restrict__ bf, const float* __restrict__ br,
                           const float* __restrict__ bcx, float* __restrict__ bcat) {
    int i = blockIdx.x * blockDim.x + threadIdx.x;   // 0..2047
    int seg = i >> 9, o = i & (N_OUT_C - 1);
    float v = 0.f;
    if (seg == 1) v = bf[o];
    else if (seg == 2) v = br[o];
    else if (seg == 3) v = bcx[o];
    bcat[i] = v;
}

// ============================================================
// bf16 WMMA GEMM:  C[M,N] = A[M,K] * W[N,K]^T + bias
// MODE 0: fp32 out, sigmoid on gate segments (col>>9 == 1 or 2), N==2048
// MODE 1: bf16 out (convert_x_layer), plain bias
// ============================================================
__device__ __forceinline__ v16bf load_frag_lds(const __bf16* __restrict__ tile,
                                               int row, int khalf) {
    const __bf16* p = tile + row * LDS_K + khalf;
    v8bf lo = *(const v8bf*)(p);        // K = khalf .. khalf+7
    v8bf hi = *(const v8bf*)(p + 16);   // K = 16+khalf .. 16+khalf+7
    return __builtin_shufflevector(lo, hi, 0,1,2,3,4,5,6,7,8,9,10,11,12,13,14,15);
}

template <int MODE>
__launch_bounds__(256, 2)
__global__ void gemm_bf16_wmma(const __bf16* __restrict__ A,   // [M,K] row-major
                               const __bf16* __restrict__ Wt,  // [N,K] row-major (weights)
                               const float*  __restrict__ bias,// [N]
                               float*        __restrict__ outF,
                               __bf16*       __restrict__ outB,
                               int M, int N, int K) {
    __shared__ __attribute__((aligned(16))) __bf16 As[2][BM * LDS_K];
    __shared__ __attribute__((aligned(16))) __bf16 Bs[2][BN * LDS_K];

    const int tid  = threadIdx.x;
    const int lane = tid & 31;
    const int wave = tid >> 5;          // 0..7 (8 wave32 waves)
    const int wm   = wave & 3;          // 4 waves along M
    const int wn   = wave >> 2;         // 2 waves along N

    const int m0 = blockIdx.x * BM;
    const int n0 = blockIdx.y * BN;

    v8f acc[2][4];
#pragma unroll
    for (int i = 0; i < 2; ++i)
#pragma unroll
        for (int j = 0; j < 4; ++j) acc[i][j] = (v8f)0.f;

    // cooperative tile loaders: 256 threads, 16B per thread-chunk, 2 chunks per tile
    const int lr = tid >> 2;            // 0..63
    const int ls = (tid & 3) * 8;       // element offset 0,8,16,24

    const __bf16* gA0 = A  + (size_t)(m0 + lr     ) * K + ls;
    const __bf16* gA1 = A  + (size_t)(m0 + lr + 64) * K + ls;
    const __bf16* gB0 = Wt + (size_t)(n0 + lr     ) * K + ls;
    const __bf16* gB1 = Wt + (size_t)(n0 + lr + 64) * K + ls;

    auto stage_tile = [&](int k0, int p) {
        copy16_to_lds(gA0 + k0, &As[p][(lr     ) * LDS_K + ls]);
        copy16_to_lds(gA1 + k0, &As[p][(lr + 64) * LDS_K + ls]);
        copy16_to_lds(gB0 + k0, &Bs[p][(lr     ) * LDS_K + ls]);
        copy16_to_lds(gB1 + k0, &Bs[p][(lr + 64) * LDS_K + ls]);
    };

    // prologue: stage first tile
    stage_tile(0, 0);
    wait_tile_copies();
    __syncthreads();

    int p = 0;
    for (int k0 = 0; k0 < K; k0 += BK) {
        if (k0 + BK < K) stage_tile(k0 + BK, p ^ 1);   // async prefetch next tile

        // fragment gather per ISA 16-bit A/B layout:
        // lanes 0-15 -> row, lane bit4 selects K half (0..7,16..23 vs 8..15,24..31)
        const int khalf = (lane >> 4) * 8;
        const int rsel  = lane & 15;

        v16bf afrag[2], bfrag[4];
#pragma unroll
        for (int i = 0; i < 2; ++i)
            afrag[i] = load_frag_lds(&As[p][0], wm * 32 + i * 16 + rsel, khalf);
#pragma unroll
        for (int j = 0; j < 4; ++j)
            bfrag[j] = load_frag_lds(&Bs[p][0], wn * 64 + j * 16 + rsel, khalf);

#pragma unroll
        for (int i = 0; i < 2; ++i)
#pragma unroll
            for (int j = 0; j < 4; ++j)
                acc[i][j] = __builtin_amdgcn_wmma_f32_16x16x32_bf16(
                    false, afrag[i], false, bfrag[j], (short)0, acc[i][j], false, false);

        wait_tile_copies();   // next tile fully in LDS
        __syncthreads();      // all waves done reading tile p
        p ^= 1;
    }

    // epilogue: C/D layout -> lane 0-15: N=lane, M=v; lane 16-31: N=lane-16, M=v+8
    const int rowBase = m0 + wm * 32;
    const int colBase = n0 + wn * 64;
    const int nlane   = lane & 15;
    const int mhi     = (lane >> 4) * 8;
#pragma unroll
    for (int i = 0; i < 2; ++i) {
#pragma unroll
        for (int j = 0; j < 4; ++j) {
            const int col = colBase + j * 16 + nlane;
            const float bv = bias[col];
            const int row0 = rowBase + i * 16 + mhi;
            if (MODE == 0) {
                const int seg = col >> 9;
                const bool sig = (seg == 1 || seg == 2);
                float* op = outF + (size_t)row0 * N + col;
#pragma unroll
                for (int v = 0; v < 8; ++v) {
                    float val = acc[i][j][v] + bv;
                    if (sig) val = 1.f / (1.f + __expf(-val));
                    op[(size_t)v * N] = val;
                }
            } else {
                __bf16* op = outB + (size_t)row0 * N + col;
#pragma unroll
                for (int v = 0; v < 8; ++v)
                    op[(size_t)v * N] = (__bf16)(acc[i][j][v] + bv);
            }
        }
    }
}

// ============================================================
// SRU elementwise scan over T. One thread per (b, o); coalesced in o.
// proj layout: [t*B+b][ x(512) | f(512) | r(512) | cx(512) ], gates pre-sigmoided.
// OUT_F32 == 0 -> bf16 h (layer 1); OUT_F32 == 1 -> fp32 h + final c (layer 2)
// ============================================================
template <int OUT_F32>
__global__ void sru_scan(const float* __restrict__ proj,
                         const float* __restrict__ c0,     // [B*N_OUT]
                         __bf16* __restrict__ h_bf,
                         float*  __restrict__ h_f,
                         float*  __restrict__ c_out) {
    const int idx = blockIdx.x * blockDim.x + threadIdx.x;  // 0..16383
    const int b = idx >> 9;
    const int o = idx & (N_OUT_C - 1);
    float c = c0[idx];
    for (int t = 0; t < T_STEPS; ++t) {
        const size_t base = ((size_t)t * BATCH + b) * N_PROJ;
        const float x  = proj[base + o];
        const float f  = proj[base + 1 * N_OUT_C + o];
        const float r  = proj[base + 2 * N_OUT_C + o];
        const float cx = proj[base + 3 * N_OUT_C + o];
        c = f * c + (1.f - f) * x;
        const float h = r * tanhf(c) + (1.f - r) * cx;
        const size_t oi = ((size_t)t * BATCH + b) * N_OUT_C + o;
        if (OUT_F32) h_f[oi]  = h;
        else         h_bf[oi] = (__bf16)h;
    }
    if (OUT_F32) c_out[idx] = c;
}

// ============================================================
// host orchestration
// ============================================================
extern "C" void kernel_launch(void* const* d_in, const int* in_sizes, int n_in,
                              void* d_out, int out_size, void* d_ws, size_t ws_size,
                              hipStream_t stream) {
    const float* xt   = (const float*)d_in[0];   // [T,B,1024]
    const float* ctf  = (const float*)d_in[1];   // [2,B,512]
    const float* W_x  = (const float*)d_in[2];   // [512,1024]
    const float* W_f  = (const float*)d_in[3];
    const float* b_f  = (const float*)d_in[4];
    const float* W_r  = (const float*)d_in[5];
    const float* b_r  = (const float*)d_in[6];
    const float* W_cx = (const float*)d_in[7];
    const float* b_cx = (const float*)d_in[8];
    const float* W_xl = (const float*)d_in[9];   // [1024,512]
    const float* b_xl = (const float*)d_in[10];  // [1024]
    float* out = (float*)d_out;                  // [T,B,512] h  ++  [B,512] c

    char* ws = (char*)d_ws;
    __bf16* Wcat = (__bf16*)(ws);                            //  4 MB  [2048,1024] bf16
    __bf16* Wxlb = (__bf16*)(ws + ((size_t)4  << 20));       //  1 MB  [1024,512]  bf16
    float*  bcat = (float* )(ws + ((size_t)5  << 20));       //  8 KB  [2048]
    __bf16* A1   = (__bf16*)(ws + ((size_t)6  << 20));       // 32 MB  xt bf16 [16384,1024]
    __bf16* A2   = (__bf16*)(ws + ((size_t)38 << 20));       // 32 MB  layer-2 input bf16
    __bf16* Ht1  = (__bf16*)(ws + ((size_t)70 << 20));       // 16 MB  layer-1 h bf16 [16384,512]
    float*  Proj = (float* )(ws + ((size_t)86 << 20));       // 128 MB [16384,2048] fp32

    const int W_ELEMS = N_OUT_C * N_IN_C;                    // 524288

    // 1) weight / input conversion to bf16
    cvt_f32_bf16<<<W_ELEMS / 256, 256, 0, stream>>>(W_x,  Wcat + 0 * (size_t)W_ELEMS, W_ELEMS);
    cvt_f32_bf16<<<W_ELEMS / 256, 256, 0, stream>>>(W_f,  Wcat + 1 * (size_t)W_ELEMS, W_ELEMS);
    cvt_f32_bf16<<<W_ELEMS / 256, 256, 0, stream>>>(W_r,  Wcat + 2 * (size_t)W_ELEMS, W_ELEMS);
    cvt_f32_bf16<<<W_ELEMS / 256, 256, 0, stream>>>(W_cx, Wcat + 3 * (size_t)W_ELEMS, W_ELEMS);
    cvt_f32_bf16<<<W_ELEMS / 256, 256, 0, stream>>>(W_xl, Wxlb, W_ELEMS);
    build_bias  <<<N_PROJ / 256,  256, 0, stream>>>(b_f, b_r, b_cx, bcat);
    cvt_f32_bf16<<<(M_TOT * N_IN_C) / 256, 256, 0, stream>>>(xt, A1, M_TOT * N_IN_C);

    dim3 blk(256);
    dim3 gProj(M_TOT / BM, N_PROJ / BN);    // 128 x 16
    dim3 gConv(M_TOT / BM, N_IN_C / BN);    // 128 x 8

    // 2) layer 1: fused 4-way projection GEMM -> scan
    gemm_bf16_wmma<0><<<gProj, blk, 0, stream>>>(A1, Wcat, bcat, Proj, nullptr,
                                                 M_TOT, N_PROJ, N_IN_C);
    sru_scan<0><<<(BATCH * N_OUT_C) / 256, 256, 0, stream>>>(Proj, ctf + 0 * BATCH * N_OUT_C,
                                                             Ht1, nullptr, nullptr);

    // 3) convert_x_layer: [16384,512] x [1024,512]^T + b_xl -> bf16 layer-2 input
    gemm_bf16_wmma<1><<<gConv, blk, 0, stream>>>(Ht1, Wxlb, b_xl, nullptr, A2,
                                                 M_TOT, N_IN_C, N_OUT_C);

    // 4) layer 2: projection GEMM -> scan, writing fp32 h and final c into d_out
    gemm_bf16_wmma<0><<<gProj, blk, 0, stream>>>(A2, Wcat, bcat, Proj, nullptr,
                                                 M_TOT, N_PROJ, N_IN_C);
    sru_scan<1><<<(BATCH * N_OUT_C) / 256, 256, 0, stream>>>(Proj, ctf + 1 * BATCH * N_OUT_C,
                                                             nullptr, out, out + (size_t)M_TOT * N_OUT_C);
}
